// MultiAttentionWithGating_4879082848603
// MI455X (gfx1250) — matrix operand order repaired and checked by reading, not compiled, not run on hardware
//
#include <hip/hip_runtime.h>
#include <hip/hip_bf16.h>
#include <math.h>

#define B_   4
#define T_   1024
#define C_   1024
#define NH_  16
#define HS_  64
#define NB_  63
#define K3C  3072

typedef __attribute__((ext_vector_type(16))) __bf16       bf16x16;
typedef __attribute__((ext_vector_type(8)))  float        f32x8;
typedef __attribute__((ext_vector_type(4)))  unsigned int u32x4;

union Frag16 { u32x4 q[2]; bf16x16 v; };

__device__ __forceinline__ unsigned short f2bf(float x) {
  unsigned int u = __float_as_uint(x);
  u = u + 0x7FFFu + ((u >> 16) & 1u);   // round-to-nearest-even
  return (unsigned short)(u >> 16);
}
__device__ __forceinline__ unsigned int pack2bf(float lo, float hi) {
  return (unsigned int)f2bf(lo) | ((unsigned int)f2bf(hi) << 16);
}

// A-fragment (16xK, 16-bit): lanes 0-15 hold K={k0..k0+7, k0+16..k0+23}, k0=0;
// lanes 16-31 same with k0=8.  Rows contiguous in LDS with `stride` shorts.
__device__ __forceinline__ bf16x16 frag_a(const unsigned short* base, int row, int stride,
                                          int kofs, int lane) {
  const unsigned short* p = base + (size_t)(row + (lane & 15)) * stride + kofs + ((lane >> 4) << 3);
  Frag16 f;
  f.q[0] = *(const u32x4*)(p);
  f.q[1] = *(const u32x4*)(p + 16);
  return f.v;
}

// B-fragment (Kx16, 16-bit), stored TRANSPOSED in LDS ([n][k]):
// lanes 0-15 hold K={kofs..kofs+7, kofs+16..kofs+23} of column n=lane, lanes 16-31 same +8.
__device__ __forceinline__ bf16x16 frag_b(const unsigned short* base, int col, int stride,
                                          int kofs, int lane) {
  const unsigned short* p = base + (size_t)(col + (lane & 15)) * stride + kofs + ((lane >> 4) << 3);
  Frag16 f;
  f.q[0] = *(const u32x4*)(p);
  f.q[1] = *(const u32x4*)(p + 16);
  return f.v;
}

#define WMMA_BF16(a, b, c) \
  __builtin_amdgcn_wmma_f32_16x16x32_bf16(false, (a), false, (b), (short)0, (c), false, false)

// ---------------------------------------------------------------------------
// GEMM: C[M,N] = A[M,K] * B[K,N] + bias[N]  (fp32 in/out, bf16 WMMA, fp32 acc)
// Block tile 128x128, BK=32. 256 threads = 8 waves (4x2), wave tile 32x64.
// LDS staged as packed bf16 pairs (b32 stores), next tiles prefetched.
// ---------------------------------------------------------------------------
__global__ __launch_bounds__(256) void gemm_bias_wmma(
    const float* __restrict__ A, const float* __restrict__ Bw,
    const float* __restrict__ bias, float* __restrict__ Cout,
    int M, int N, int Kd)
{
  __shared__ __align__(16) unsigned int AsU[128 * 16];   // [m][k/2]
  __shared__ __align__(16) unsigned int BtU[128 * 16];   // [n][k/2]

  const int tid  = threadIdx.x;
  const int lane = tid & 31;
  const int wid  = tid >> 5;
  const int m0 = blockIdx.y * 128;
  const int n0 = blockIdx.x * 128;
  const int wm = (wid & 3) * 32;
  const int wn = (wid >> 2) * 64;

  f32x8 acc[2][4] = {};

  for (int k0 = 0; k0 < Kd; k0 += 32) {
#pragma unroll
    for (int i = 0; i < 8; ++i) {                  // A tile 128x32 -> 2048 pairs
      int li = tid + 256 * i;
      int r = li >> 4, cp = li & 15;
      const float2 a2 = *(const float2*)(A + (size_t)(m0 + r) * Kd + k0 + 2 * cp);
      AsU[r * 16 + cp] = pack2bf(a2.x, a2.y);
    }
#pragma unroll
    for (int i = 0; i < 8; ++i) {                  // B tile 32x128 -> transposed pairs
      int li = tid + 256 * i;
      int n = li & 127, kp = li >> 7;              // kp in 0..15
      float b0 = Bw[(size_t)(k0 + 2 * kp)     * N + n0 + n];
      float b1 = Bw[(size_t)(k0 + 2 * kp + 1) * N + n0 + n];
      BtU[n * 16 + kp] = pack2bf(b0, b1);
    }
    if (k0 + 32 < Kd) {                            // speculative prefetch of next K-tile
      __builtin_prefetch(A  + (size_t)(m0 + (tid >> 1)) * Kd + k0 + 32 + (tid & 1) * 16, 0, 1);
      __builtin_prefetch(Bw + (size_t)(k0 + 32 + (tid >> 6)) * N + n0 + ((tid & 63) << 1), 0, 1);
    }
    __syncthreads();

    bf16x16 a0 = frag_a((const unsigned short*)AsU, wm + 0,  32, 0, lane);
    bf16x16 a1 = frag_a((const unsigned short*)AsU, wm + 16, 32, 0, lane);
#pragma unroll
    for (int ni = 0; ni < 4; ++ni) {
      bf16x16 bN = frag_b((const unsigned short*)BtU, wn + ni * 16, 32, 0, lane);
      acc[0][ni] = WMMA_BF16(a0, bN, acc[0][ni]);
      acc[1][ni] = WMMA_BF16(a1, bN, acc[1][ni]);
    }
    __syncthreads();
  }

#pragma unroll
  for (int mi = 0; mi < 2; ++mi)
#pragma unroll
    for (int ni = 0; ni < 4; ++ni) {
      int col = n0 + wn + ni * 16 + (lane & 15);
      float bv = bias[col];
#pragma unroll
      for (int r = 0; r < 8; ++r) {
        int row = m0 + wm + mi * 16 + r + ((lane >> 4) << 3);
        Cout[(size_t)row * N + col] = acc[mi][ni][r] + bv;
      }
    }
}

// ---------------------------------------------------------------------------
// Flash-style causal attention for one (b,h) and a 64-query tile.
// 128 threads = 4 waves, wave w owns query rows w*16..w*16+15.
// ---------------------------------------------------------------------------
__global__ __launch_bounds__(128) void local_attn_kernel(
    const float* __restrict__ qkv, float* __restrict__ outLocal)
{
  __shared__ __align__(16) unsigned int QsU[64 * 32];   // [q][hs/2]
  __shared__ __align__(16) unsigned int KsU[64 * 32];   // [key][hs/2]
  __shared__ __align__(16) unsigned int VtU[64 * 32];   // [hs][key/2]
  __shared__ __align__(16) unsigned int PsU[64 * 32];   // [q][key/2]
  __shared__ __align__(16) float        Ss[64 * 64];    // [q][key]
  __shared__ float corr[64];

  const int tid  = threadIdx.x;
  const int lane = tid & 31;
  const int wid  = tid >> 5;
  const int bh = blockIdx.x;
  const int b = bh >> 4, h = bh & 15;
  const int qtile = blockIdx.y;
  const int qbase = qtile * 64;

#pragma unroll
  for (int i = 0; i < 16; ++i) {                 // load Q 64x64 (packed pairs)
    int li = tid + 128 * i;
    int r = li >> 5, cp = li & 31;
    const float2 q2 = *(const float2*)(qkv + ((size_t)b * T_ + qbase + r) * K3C + h * HS_ + 2 * cp);
    QsU[r * 32 + cp] = pack2bf(q2.x, q2.y);
  }

  f32x8 o[4] = {};
  float rm = -1e30f, rl = 0.f;

  for (int j = 0; j <= qtile; ++j) {
    const int kb = j * 64;
    __syncthreads();                              // protect K/V/P reuse (and Q on iter 0)
#pragma unroll
    for (int i = 0; i < 16; ++i) {                // K tile, packed along hs
      int li = tid + 128 * i;
      int r = li >> 5, cp = li & 31;
      const float2 k2 = *(const float2*)(qkv + ((size_t)b * T_ + kb + r) * K3C + C_ + h * HS_ + 2 * cp);
      KsU[r * 32 + cp] = pack2bf(k2.x, k2.y);
    }
#pragma unroll
    for (int i = 0; i < 16; ++i) {                // V tile transposed, packed along key
      int li = tid + 128 * i;
      int hs = li & 63, kp = li >> 6;             // kp in 0..31
      size_t base = ((size_t)b * T_ + kb + 2 * kp) * K3C + 2 * C_ + h * HS_ + hs;
      VtU[hs * 32 + kp] = pack2bf(qkv[base], qkv[base + K3C]);
    }
    if (j < qtile) {                              // prefetch next K/V tile
      __builtin_prefetch(qkv + ((size_t)b * T_ + kb + 64 + (tid & 63)) * K3C + C_     + h * HS_, 0, 1);
      __builtin_prefetch(qkv + ((size_t)b * T_ + kb + 64 + (tid & 63)) * K3C + 2 * C_ + h * HS_, 0, 1);
    }
    __syncthreads();

    // S = Q * K^T   (contraction over hs=64 -> 2 WMMA k-steps)
    bf16x16 aq0 = frag_a((const unsigned short*)QsU, wid * 16, 64, 0,  lane);
    bf16x16 aq1 = frag_a((const unsigned short*)QsU, wid * 16, 64, 32, lane);
#pragma unroll
    for (int ni = 0; ni < 4; ++ni) {
      f32x8 s = {};
      s = WMMA_BF16(aq0, frag_b((const unsigned short*)KsU, ni * 16, 64, 0,  lane), s);
      s = WMMA_BF16(aq1, frag_b((const unsigned short*)KsU, ni * 16, 64, 32, lane), s);
      int coln = ni * 16 + (lane & 15);
#pragma unroll
      for (int r = 0; r < 8; ++r)
        Ss[(wid * 16 + r + ((lane >> 4) << 3)) * 64 + coln] = s[r];
    }
    __syncthreads();

    // per-row online softmax (one thread per query row)
    if (tid < 64) {
      const int row = tid;
      const int q = qbase + row;
      int kv = q - kb + 1; if (kv > 64) kv = 64;
      const float scale = 0.125f;
      float tmax = -1e30f;
      for (int n = 0; n < kv; ++n) {
        float sv = Ss[row * 64 + n] * scale;
        Ss[row * 64 + n] = sv;
        tmax = fmaxf(tmax, sv);
      }
      float newm = fmaxf(rm, tmax);
      float cf = __expf(rm - newm);
      float lsum = 0.f;
      for (int n2 = 0; n2 < 32; ++n2) {
        int n = 2 * n2;
        float e0 = 0.f, e1 = 0.f;
        if (n < kv)     { e0 = __expf(Ss[row * 64 + n]     - newm); lsum += e0; }
        if (n + 1 < kv) { e1 = __expf(Ss[row * 64 + n + 1] - newm); lsum += e1; }
        PsU[row * 32 + n2] = pack2bf(e0, e1);
      }
      rl = rl * cf + lsum;
      rm = newm;
      corr[row] = cf;
    }
    __syncthreads();

    // rescale accumulators, then O += P * V
    float cfs[8];
#pragma unroll
    for (int r = 0; r < 8; ++r)
      cfs[r] = corr[wid * 16 + r + ((lane >> 4) << 3)];
#pragma unroll
    for (int ni = 0; ni < 4; ++ni)
#pragma unroll
      for (int r = 0; r < 8; ++r)
        o[ni][r] *= cfs[r];

    bf16x16 ap0 = frag_a((const unsigned short*)PsU, wid * 16, 64, 0,  lane);
    bf16x16 ap1 = frag_a((const unsigned short*)PsU, wid * 16, 64, 32, lane);
#pragma unroll
    for (int ni = 0; ni < 4; ++ni) {
      o[ni] = WMMA_BF16(ap0, frag_b((const unsigned short*)VtU, ni * 16, 64, 0,  lane), o[ni]);
      o[ni] = WMMA_BF16(ap1, frag_b((const unsigned short*)VtU, ni * 16, 64, 32, lane), o[ni]);
    }
  }

  __syncthreads();
  if (tid < 64) corr[tid] = 1.0f / rl;
  __syncthreads();

  float cfs[8];
#pragma unroll
  for (int r = 0; r < 8; ++r)
    cfs[r] = corr[wid * 16 + r + ((lane >> 4) << 3)];
#pragma unroll
  for (int ni = 0; ni < 4; ++ni) {
    int colC = h * HS_ + ni * 16 + (lane & 15);
#pragma unroll
    for (int r = 0; r < 8; ++r) {
      int row = qbase + wid * 16 + r + ((lane >> 4) << 3);
      outLocal[((size_t)b * T_ + row) * C_ + colC] = o[ni][r] * cfs[r];
    }
  }
}

// ---------------------------------------------------------------------------
// Conv1d(K=32, stride=16) + LN + exact GELU, applied to k or v (blockIdx.y).
// ---------------------------------------------------------------------------
__global__ __launch_bounds__(64) void compress_kernel(
    const float* __restrict__ qkv, const float* __restrict__ conv_w,
    const float* __restrict__ conv_b, const float* __restrict__ lnw,
    const float* __restrict__ lnb, float* __restrict__ kc, float* __restrict__ vc)
{
  __shared__ float patch[64 * 32];
  __shared__ float red[64];
  const int tid = threadIdx.x;
  const int bidx = blockIdx.x;
  const int blk = bidx % NB_;
  const int bh = bidx / NB_;
  const int b = bh >> 4, h = bh & 15;
  const int srcOff = (blockIdx.y == 0) ? C_ : 2 * C_;
  float* dst = (blockIdx.y == 0) ? kc : vc;
  const int t0 = blk * 16;

#pragma unroll
  for (int i = 0; i < 32; ++i) {
    int li = tid + 64 * i;
    int hsin = li >> 5, kk = li & 31;
    patch[hsin * 32 + kk] = qkv[((size_t)b * T_ + t0 + kk) * K3C + srcOff + h * HS_ + hsin];
  }
  __syncthreads();

  float acc = conv_b[tid];
  for (int hsin = 0; hsin < 64; ++hsin) {
    const float* wr = conv_w + ((size_t)tid * 64 + hsin) * 32;
#pragma unroll
    for (int kk = 0; kk < 32; ++kk)
      acc += patch[hsin * 32 + kk] * wr[kk];
  }

  red[tid] = acc; __syncthreads();
  for (int s = 32; s > 0; s >>= 1) { if (tid < s) red[tid] += red[tid + s]; __syncthreads(); }
  float mean = red[0] * (1.0f / 64.0f);
  __syncthreads();
  float d = acc - mean;
  red[tid] = d * d; __syncthreads();
  for (int s = 32; s > 0; s >>= 1) { if (tid < s) red[tid] += red[tid + s]; __syncthreads(); }
  float var = red[0] * (1.0f / 64.0f);

  float y = d * rsqrtf(var + 1e-5f) * lnw[tid] + lnb[tid];
  float g = 0.5f * y * (1.0f + erff(y * 0.7071067811865475f));
  dst[((size_t)bh * NB_ + blk) * HS_ + tid] = g;
}

// ---------------------------------------------------------------------------
// Row LayerNorm over C=1024 (one 256-thread block per row).
// ---------------------------------------------------------------------------
__global__ __launch_bounds__(256) void ln_row_kernel(
    const float* __restrict__ in, const float* __restrict__ w,
    const float* __restrict__ bb, float* __restrict__ out)
{
  __shared__ float red[256];
  const int tid = threadIdx.x;
  const size_t row = blockIdx.x;
  const float* xr = in + row * C_;
  float v[4]; float s = 0.f;
#pragma unroll
  for (int i = 0; i < 4; ++i) { v[i] = xr[tid + 256 * i]; s += v[i]; }
  red[tid] = s; __syncthreads();
  for (int t = 128; t > 0; t >>= 1) { if (tid < t) red[tid] += red[tid + t]; __syncthreads(); }
  float mean = red[0] * (1.0f / 1024.0f);
  __syncthreads();
  float ss = 0.f;
#pragma unroll
  for (int i = 0; i < 4; ++i) { float d = v[i] - mean; ss += d * d; }
  red[tid] = ss; __syncthreads();
  for (int t = 128; t > 0; t >>= 1) { if (tid < t) red[tid] += red[tid + t]; __syncthreads(); }
  float rstd = rsqrtf(red[0] * (1.0f / 1024.0f) + 1e-5f);
  float* yr = out + row * C_;
#pragma unroll
  for (int i = 0; i < 4; ++i) {
    int c = tid + 256 * i;
    yr[c] = (v[i] - mean) * rstd * w[c] + bb[c];
  }
}

// ---------------------------------------------------------------------------
// Compressed attention: 63 keys, mask j <= t.  One 64-thread block per (b,h,t).
// ---------------------------------------------------------------------------
__global__ __launch_bounds__(64) void comp_attn_kernel(
    const float* __restrict__ qkv, const float* __restrict__ kc,
    const float* __restrict__ vc, float* __restrict__ outComp)
{
  __shared__ float qs[64];
  __shared__ float sc[64];
  __shared__ float red2[2];
  const int tid = threadIdx.x;
  const int idx = blockIdx.x;
  const int t = idx & (T_ - 1);
  const int bh = idx >> 10;
  const int b = bh >> 4, h = bh & 15;

  qs[tid] = qkv[((size_t)b * T_ + t) * K3C + h * HS_ + tid];
  __syncthreads();
  const int jmax = t < 62 ? t : 62;
  const int cnt = jmax + 1;

  if (tid < cnt) {
    const float* kr = kc + ((size_t)bh * NB_ + tid) * HS_;
    float s = 0.f;
#pragma unroll
    for (int d = 0; d < 64; ++d) s += qs[d] * kr[d];
    sc[tid] = s * 0.125f;
  }
  __syncthreads();
  if (tid == 0) {
    float mx = -1e30f;
    for (int n = 0; n < cnt; ++n) mx = fmaxf(mx, sc[n]);
    red2[0] = mx;
  }
  __syncthreads();
  float mx = red2[0];
  if (tid < cnt) sc[tid] = __expf(sc[tid] - mx);
  __syncthreads();
  if (tid == 0) {
    float l = 0.f;
    for (int n = 0; n < cnt; ++n) l += sc[n];
    red2[1] = 1.0f / l;
  }
  __syncthreads();
  float linv = red2[1];
  float acc = 0.f;
  for (int n = 0; n < cnt; ++n)
    acc += sc[n] * vc[((size_t)bh * NB_ + n) * HS_ + tid];
  outComp[((size_t)b * T_ + t) * C_ + h * HS_ + tid] = acc * linv;
}

// ---------------------------------------------------------------------------
// Gating MLP stage 1: hpre[b,:] = feats[b,:] @ gate_w1 + b1  (feats = concat)
// ---------------------------------------------------------------------------
__global__ __launch_bounds__(256) void gate1_kernel(
    const float* __restrict__ lnorm, const float* __restrict__ comp,
    const float* __restrict__ w1, const float* __restrict__ b1,
    float* __restrict__ hpre)
{
  const int b = blockIdx.x, tid = threadIdx.x;
  const float* f1 = lnorm + ((size_t)b * T_ + (T_ - 1)) * C_;
  const float* f2 = comp  + ((size_t)b * T_ + (T_ - 1)) * C_;
  float acc[4] = { b1[tid], b1[tid + 256], b1[tid + 512], b1[tid + 768] };
  for (int k = 0; k < 2 * C_; ++k) {
    float f = (k < C_) ? f1[k] : f2[k - C_];
    const float* wr = w1 + (size_t)k * C_;
    acc[0] += f * wr[tid];
    acc[1] += f * wr[tid + 256];
    acc[2] += f * wr[tid + 512];
    acc[3] += f * wr[tid + 768];
  }
  hpre[b * C_ + tid]       = acc[0];
  hpre[b * C_ + tid + 256] = acc[1];
  hpre[b * C_ + tid + 512] = acc[2];
  hpre[b * C_ + tid + 768] = acc[3];
}

// ---------------------------------------------------------------------------
// Gating stage 2: LN + relu + @gate_w2 + softmax -> gates[b][0..1]
// ---------------------------------------------------------------------------
__global__ __launch_bounds__(256) void gate2_kernel(
    const float* __restrict__ hpre, const float* __restrict__ glw,
    const float* __restrict__ glb, const float* __restrict__ w2,
    const float* __restrict__ b2, float* __restrict__ gates)
{
  __shared__ float red[256];
  const int b = blockIdx.x, tid = threadIdx.x;
  const float* xr = hpre + b * C_;
  float v[4]; float s = 0.f;
#pragma unroll
  for (int i = 0; i < 4; ++i) { v[i] = xr[tid + 256 * i]; s += v[i]; }
  red[tid] = s; __syncthreads();
  for (int t = 128; t > 0; t >>= 1) { if (tid < t) red[tid] += red[tid + t]; __syncthreads(); }
  float mean = red[0] * (1.0f / 1024.0f);
  __syncthreads();
  float ss = 0.f;
#pragma unroll
  for (int i = 0; i < 4; ++i) { float d = v[i] - mean; ss += d * d; }
  red[tid] = ss; __syncthreads();
  for (int t = 128; t > 0; t >>= 1) { if (tid < t) red[tid] += red[tid + t]; __syncthreads(); }
  float rstd = rsqrtf(red[0] * (1.0f / 1024.0f) + 1e-5f);
  __syncthreads();

  float z0p = 0.f, z1p = 0.f;
#pragma unroll
  for (int i = 0; i < 4; ++i) {
    int c = tid + 256 * i;
    float hv = fmaxf(0.f, (v[i] - mean) * rstd * glw[c] + glb[c]);
    z0p += hv * w2[c * 2 + 0];
    z1p += hv * w2[c * 2 + 1];
  }
  red[tid] = z0p; __syncthreads();
  for (int t = 128; t > 0; t >>= 1) { if (tid < t) red[tid] += red[tid + t]; __syncthreads(); }
  float z0 = red[0]; __syncthreads();
  red[tid] = z1p; __syncthreads();
  for (int t = 128; t > 0; t >>= 1) { if (tid < t) red[tid] += red[tid + t]; __syncthreads(); }
  float z1 = red[0];
  if (tid == 0) {
    z0 += b2[0]; z1 += b2[1];
    float mx = fmaxf(z0, z1);
    float e0 = __expf(z0 - mx), e1 = __expf(z1 - mx);
    float inv = 1.0f / (e0 + e1);
    gates[b * 2 + 0] = e0 * inv;
    gates[b * 2 + 1] = e1 * inv;
  }
}

__global__ __launch_bounds__(256) void combine_kernel(
    const float* __restrict__ lnorm, const float* __restrict__ comp,
    const float* __restrict__ gates, float* __restrict__ comb)
{
  size_t idx = (size_t)blockIdx.x * 256 + threadIdx.x;
  int b = (int)(idx >> 20);   // T_*C_ = 2^20
  float g0 = gates[b * 2], g1 = gates[b * 2 + 1];
  comb[idx] = g0 * lnorm[idx] + g1 * comp[idx];
}

// ---------------------------------------------------------------------------
// Workspace layout (float offsets)
// ---------------------------------------------------------------------------
#define OFF_QKV   0ull
#define OFF_KC    (OFF_QKV   + 12582912ull)   // B*T*3C
#define OFF_VC    (OFF_KC    + 258048ull)     // B*NH*63*64
#define OFF_LOCAL (OFF_VC    + 258048ull)
#define OFF_LNORM (OFF_LOCAL + 4194304ull)    // B*T*C
#define OFF_COMP  (OFF_LNORM + 4194304ull)
#define OFF_COMB  (OFF_COMP  + 4194304ull)
#define OFF_HPRE  (OFF_COMB  + 4194304ull)
#define OFF_GATES (OFF_HPRE  + 4096ull)

extern "C" void kernel_launch(void* const* d_in, const int* in_sizes, int n_in,
                              void* d_out, int out_size, void* d_ws, size_t ws_size,
                              hipStream_t stream) {
  (void)in_sizes; (void)n_in; (void)out_size; (void)ws_size;
  const float* x         = (const float*)d_in[0];
  const float* c_attn_w  = (const float*)d_in[1];
  const float* c_attn_b  = (const float*)d_in[2];
  const float* c_proj_w  = (const float*)d_in[3];
  const float* c_proj_b  = (const float*)d_in[4];
  const float* conv_w    = (const float*)d_in[5];
  const float* conv_b    = (const float*)d_in[6];
  const float* comp_ln_w = (const float*)d_in[7];
  const float* comp_ln_b = (const float*)d_in[8];
  const float* ln_loc_w  = (const float*)d_in[9];
  const float* ln_loc_b  = (const float*)d_in[10];
  const float* gate_w1   = (const float*)d_in[11];
  const float* gate_b1   = (const float*)d_in[12];
  const float* gate_ln_w = (const float*)d_in[13];
  const float* gate_ln_b = (const float*)d_in[14];
  const float* gate_w2   = (const float*)d_in[15];
  const float* gate_b2   = (const float*)d_in[16];

  float* ws    = (float*)d_ws;
  float* qkv   = ws + OFF_QKV;
  float* kc    = ws + OFF_KC;
  float* vc    = ws + OFF_VC;
  float* local = ws + OFF_LOCAL;
  float* lnorm = ws + OFF_LNORM;
  float* comp  = ws + OFF_COMP;
  float* comb  = ws + OFF_COMB;
  float* hpre  = ws + OFF_HPRE;
  float* gates = ws + OFF_GATES;
  float* out   = (float*)d_out;

  // 1) qkv = x @ c_attn_w + b       (4096 x 3072, K=1024)
  gemm_bias_wmma<<<dim3(K3C / 128, (B_ * T_) / 128), 256, 0, stream>>>(
      x, c_attn_w, c_attn_b, qkv, B_ * T_, K3C, C_);
  // 2) compress k and v
  compress_kernel<<<dim3(B_ * NH_ * NB_, 2), 64, 0, stream>>>(
      qkv, conv_w, conv_b, comp_ln_w, comp_ln_b, kc, vc);
  // 3) local causal attention (flash, WMMA)
  local_attn_kernel<<<dim3(B_ * NH_, T_ / 64), 128, 0, stream>>>(qkv, local);
  // 4) local_norm = LN(local)
  ln_row_kernel<<<B_ * T_, 256, 0, stream>>>(local, ln_loc_w, ln_loc_b, lnorm);
  // 5) compressed attention
  comp_attn_kernel<<<B_ * NH_ * T_, 64, 0, stream>>>(qkv, kc, vc, comp);
  // 6) gating MLP
  gate1_kernel<<<B_, 256, 0, stream>>>(lnorm, comp, gate_w1, gate_b1, hpre);
  gate2_kernel<<<B_, 256, 0, stream>>>(hpre, gate_ln_w, gate_ln_b, gate_w2, gate_b2, gates);
  // 7) combined = g0*local_norm + g1*comp
  combine_kernel<<<(B_ * T_ * C_) / 256, 256, 0, stream>>>(lnorm, comp, gates, comb);
  // 8) out = combined @ c_proj_w + b    (4096 x 1024, K=1024)
  gemm_bias_wmma<<<dim3(C_ / 128, (B_ * T_) / 128), 256, 0, stream>>>(
      comb, c_proj_w, c_proj_b, out, B_ * T_, C_, C_);
}
